// FlashAttentionLayer_35897336660137
// MI455X (gfx1250) — matrix operand order repaired and checked by reading, not compiled, not run on hardware
//
#include <hip/hip_runtime.h>
#include <math.h>

// Flash attention (key-padding mask) for MI455X / gfx1250, wave32 WMMA.
// B=4, S=2048, H=16, D=128.  fp32 in/out, f16 WMMA with f32 accumulate.
// v2: LDS pitch padding (bank-conflict free fragment loads) + 2-stage
//     software pipeline overlapping global K/V loads with WMMA compute.

#define BSZ   4
#define SEQ   2048
#define NH    16
#define HD    128
#define QT    128          // query rows per block (8 waves x 16 rows)
#define KTL   32           // keys per tile
#define NWAVE 8

// LDS pitches (in halves) chosen so row-stride in 32-bit words is not a
// multiple of the 64-bank period: 136h = 68 words (bank step 4),
// 40h = 20 words (16 distinct banks over 16 rows). All offsets 16B aligned.
#define KPITCH 136
#define VPITCH 40
#define PPITCH 40

static constexpr float SCALEF = 0.08838834764831845f; // 1/sqrt(128)

typedef __attribute__((ext_vector_type(16))) _Float16 v16h;
typedef __attribute__((ext_vector_type(8)))  _Float16 v8h;
typedef __attribute__((ext_vector_type(8)))  float    v8f;

// Build a 16-half WMMA fragment from two 16B LDS chunks (16B-aligned).
__device__ __forceinline__ v16h ld_frag2(const _Float16* p0, const _Float16* p1) {
    v8h lo = *(const v8h*)p0;
    v8h hi = *(const v8h*)p1;
    v16h r;
    #pragma unroll
    for (int i = 0; i < 8; ++i) { r[i] = lo[i]; r[8 + i] = hi[i]; }
    return r;
}

__global__ __launch_bounds__(256, 1)
void fa_fwd_wmma(const float* __restrict__ q,
                 const float* __restrict__ kk,
                 const float* __restrict__ vv,
                 const int*   __restrict__ msk,
                 float* __restrict__ out)
{
    // Double-buffered K (row-major f16) and V (transposed f16) + per-wave P.
    __shared__ _Float16 lds_k [2][KTL][KPITCH];   // ~17 KB
    __shared__ _Float16 lds_vT[2][HD][VPITCH];    // ~20 KB
    __shared__ _Float16 lds_p [NWAVE][16][PPITCH];// ~10 KB

    const int tid  = threadIdx.x;
    const int wave = tid >> 5;
    const int lane = tid & 31;
    const int half = lane >> 4;    // 0: rows 0-7 / k-lo groups, 1: rows 8-15 / k-hi
    const int ln   = lane & 15;    // column / row index within 16

    const int bh = blockIdx.y;
    const int b  = bh / NH;
    const int h  = bh % NH;
    const int qbase = blockIdx.x * QT;

    // ---------------- Load Q stripe as 4 A-fragments (16x32 f16 each) --------
    const int qrow = qbase + wave * 16 + ln;      // A-layout: M = lane%16
    const float* qp = q + (((size_t)b * SEQ + qrow) * NH + h) * HD;
    v16h qa[4];
    #pragma unroll
    for (int c = 0; c < 4; ++c) {
        const int base = c * 32 + half * 16;
        #pragma unroll
        for (int i = 0; i < 8; ++i) {
            qa[c][i]     = (_Float16)qp[base + i];        // K = base..base+7
            qa[c][8 + i] = (_Float16)qp[base + 8 + i];    // K = base+8..base+15
        }
    }

    // ---------------- Accumulators (8 d-tiles of 16x16 f32) + softmax state --
    v8f acc[8];
    #pragma unroll
    for (int t = 0; t < 8; ++t)
        #pragma unroll
        for (int i = 0; i < 8; ++i) acc[t][i] = 0.0f;
    float rm[8], rl[8];
    #pragma unroll
    for (int r = 0; r < 8; ++r) { rm[r] = -INFINITY; rl[r] = 0.0f; }

    // ---------------- Pipelined K/V tile loader -------------------------------
    const int row = tid >> 3;          // 0..31 key row within tile
    const int d0  = (tid & 7) * 16;    // 16 contiguous head dims
    const size_t kvbase = (((size_t)b * SEQ + row) * NH + h) * HD + d0;
    const size_t tilestep = (size_t)KTL * NH * HD;   // advance KTL keys

    float kreg[16], vreg[16];
    auto load_tile = [&](int kt_) {
        const float* kp = kk + kvbase + (size_t)kt_ * tilestep;
        const float* vp = vv + kvbase + (size_t)kt_ * tilestep;
        #pragma unroll
        for (int i = 0; i < 16; ++i) kreg[i] = kp[i];
        #pragma unroll
        for (int i = 0; i < 16; ++i) vreg[i] = vp[i];
    };
    auto store_tile = [&](int buf) {
        #pragma unroll
        for (int i = 0; i < 16; ++i) lds_k[buf][row][d0 + i]  = (_Float16)kreg[i];
        #pragma unroll
        for (int i = 0; i < 16; ++i) lds_vT[buf][d0 + i][row] = (_Float16)vreg[i];
    };

    const int ntiles = SEQ / KTL;   // 64

    // prologue: stage tile 0
    load_tile(0);
    store_tile(0);
    __syncthreads();

    for (int kt = 0; kt < ntiles; ++kt) {
        const int key0 = kt * KTL;
        const int cur  = kt & 1;

        // Issue next tile's global loads now; they retire behind the WMMAs.
        if (kt + 1 < ntiles) load_tile(kt + 1);
        if (kt + 2 < ntiles) {   // keep L2/WGP$ warm two tiles ahead
            __builtin_prefetch(kk + kvbase + (size_t)(kt + 2) * tilestep, 0, 1);
            __builtin_prefetch(vv + kvbase + (size_t)(kt + 2) * tilestep, 0, 1);
        }

        // ---- S = Q * K^T : two 16x16 score tiles, 4 k-chunks each -----------
        v8f s0 = {}, s1 = {};
        #pragma unroll
        for (int c = 0; c < 4; ++c) {
            // B-layout: lane column n = ln selects key; half selects k 0-15/16-31
            const _Float16* kb0 = &lds_k[cur][ln][c * 32 + half * 16];
            v16h b0 = ld_frag2(kb0, kb0 + 8);
            s0 = __builtin_amdgcn_wmma_f32_16x16x32_f16(false, qa[c], false, b0,
                                                        (short)0, s0, false, false);
            const _Float16* kb1 = &lds_k[cur][16 + ln][c * 32 + half * 16];
            v16h b1 = ld_frag2(kb1, kb1 + 8);
            s1 = __builtin_amdgcn_wmma_f32_16x16x32_f16(false, qa[c], false, b1,
                                                        (short)0, s1, false, false);
        }

        // ---- mask + scale + online softmax ----------------------------------
        const int mk0 = msk[b * SEQ + key0 + ln];
        const int mk1 = msk[b * SEQ + key0 + 16 + ln];
        float p0[8], p1[8];
        #pragma unroll
        for (int r = 0; r < 8; ++r) {
            p0[r] = mk0 ? s0[r] * SCALEF : -1.0e30f;
            p1[r] = mk1 ? s1[r] * SCALEF : -1.0e30f;
        }
        #pragma unroll
        for (int r = 0; r < 8; ++r) {
            // row max across the 16 lanes holding this row's columns
            float m = fmaxf(p0[r], p1[r]);
            m = fmaxf(m, __shfl_xor(m, 1));
            m = fmaxf(m, __shfl_xor(m, 2));
            m = fmaxf(m, __shfl_xor(m, 4));
            m = fmaxf(m, __shfl_xor(m, 8));
            const float mnew  = fmaxf(rm[r], m);
            const float alpha = __expf(rm[r] - mnew);
            rm[r] = mnew;
            p0[r] = __expf(p0[r] - mnew);
            p1[r] = __expf(p1[r] - mnew);
            float rs = p0[r] + p1[r];
            rs += __shfl_xor(rs, 1);
            rs += __shfl_xor(rs, 2);
            rs += __shfl_xor(rs, 4);
            rs += __shfl_xor(rs, 8);
            rl[r] = rl[r] * alpha + rs;
            #pragma unroll
            for (int t = 0; t < 8; ++t) acc[t][r] *= alpha;   // rescale O rows
        }

        // ---- D-layout -> A-layout for P via per-wave LDS scratch ------------
        #pragma unroll
        for (int r = 0; r < 8; ++r) {
            lds_p[wave][r + half * 8][ln]      = (_Float16)p0[r];
            lds_p[wave][r + half * 8][16 + ln] = (_Float16)p1[r];
        }
        __builtin_amdgcn_wave_barrier();   // wave-internal; DS ops are in order
        const _Float16* prow = &lds_p[wave][ln][0];
        v16h pa = ld_frag2(prow + half * 16, prow + half * 16 + 8);

        // ---- O += P * V : 8 output d-tiles ----------------------------------
        #pragma unroll
        for (int t = 0; t < 8; ++t) {
            // B-layout: n = ln -> d = t*16+ln; half selects key 0-15 / 16-31
            const _Float16* vb = &lds_vT[cur][t * 16 + ln][half * 16];
            v16h bv = ld_frag2(vb, vb + 8);
            acc[t] = __builtin_amdgcn_wmma_f32_16x16x32_f16(false, pa, false, bv,
                                                            (short)0, acc[t], false, false);
        }

        // ---- stage next tile into the other buffer --------------------------
        if (kt + 1 < ntiles) store_tile(1 - cur);
        __syncthreads();
    }

    // ---------------- normalize and store ------------------------------------
    #pragma unroll
    for (int r = 0; r < 8; ++r) {
        const float inv = 1.0f / rl[r];
        const int qr = qbase + wave * 16 + r + half * 8;
        float* op = out + (((size_t)b * SEQ + qr) * NH + h) * HD;
        #pragma unroll
        for (int t = 0; t < 8; ++t) op[t * 16 + ln] = acc[t][r] * inv;
    }
}

extern "C" void kernel_launch(void* const* d_in, const int* in_sizes, int n_in,
                              void* d_out, int out_size, void* d_ws, size_t ws_size,
                              hipStream_t stream) {
    const float* q  = (const float*)d_in[0];
    const float* k  = (const float*)d_in[1];
    const float* v  = (const float*)d_in[2];
    const int*   m  = (const int*)d_in[3];
    float* out = (float*)d_out;

    dim3 grid(SEQ / QT, BSZ * NH);   // (16, 64)
    dim3 block(256);                 // 8 waves
    hipLaunchKernelGGL(fa_fwd_wmma, grid, block, 0, stream, q, k, v, m, out);
}
